// GraphEncoder_36782099923299
// MI455X (gfx1250) — matrix operand order repaired and checked by reading, not compiled, not run on hardware
//
#include <hip/hip_runtime.h>
#include <hip/hip_bf16.h>

typedef __bf16 bf16_t;
typedef __attribute__((ext_vector_type(16))) __bf16 v16bf;
typedef __attribute__((ext_vector_type(8)))  __bf16 v8bf;
typedef __attribute__((ext_vector_type(8)))  float  v8f;

#define N_NODES 10000
#define N_EDGES 80000
#define ET      (N_EDGES + N_NODES)   // edges + self-loops
#define IN_DIM  768
#define HID     512
#define HEADS   4
#define HDC     (HEADS * HID)         // 2048
#define OUT_DIM 768

// ---------------- elementwise / conversion ----------------

__global__ void k_f32_to_bf16(const float* __restrict__ in, bf16_t* __restrict__ out, int n) {
  int i = blockIdx.x * blockDim.x + threadIdx.x;
  if (i < n) out[i] = (bf16_t)in[i];
}

// in: [K][Ncols] f32 row-major  ->  out: [Ncols][K] bf16 row-major (B^T for GEMM)
__global__ void k_transpose_bf16(const float* __restrict__ in, bf16_t* __restrict__ out,
                                 int K, int Ncols) {
  int i = blockIdx.x * blockDim.x + threadIdx.x;
  if (i < K * Ncols) {
    int k = i / Ncols, n = i - k * Ncols;
    out[(size_t)n * K + k] = (bf16_t)in[i];
  }
}

__global__ void k_fill(float* __restrict__ p, float v, int n) {
  int i = blockIdx.x * blockDim.x + threadIdx.x;
  if (i < n) p[i] = v;
}

// ---------------- WMMA GEMM:  C[M,N] = A[M,K] * Bt[N,K]^T  (bf16 in, f32 out) ----------------
// One wave computes a 16(M) x 64(N) tile: 4 f32 accumulators, A-fragment reused 4x.

__device__ inline v16bf load_afrag(const bf16_t* p) {
  // A 16x32 fragment in this code's contiguous k order (B matches): lanes 0-15 hold K = {k0..k0+15},
  // lanes 16-31 hold K = {k0+16..k0+31}  (caller passes p already half-offset)
  union { v16bf v; struct { v8bf lo, hi; } s; } u;
  u.s.lo = *(const v8bf*)(p);
  u.s.hi = *(const v8bf*)(p + 8);
  return u.v;
}

__global__ __launch_bounds__(32) void k_gemm_bf16_wmma(
    const bf16_t* __restrict__ A,   // [M,K] row-major bf16
    const bf16_t* __restrict__ Bt,  // [N,K] row-major bf16 (pre-transposed B)
    float* __restrict__ C,          // [M,N] row-major f32
    int M, int N, int K)
{
  const int lane = threadIdx.x & 31;
  const int hv   = lane >> 4;   // half-wave: 0 or 1
  const int r    = lane & 15;
  const int m0   = blockIdx.x * 16;
  const int n0   = blockIdx.y * 64;

  const bf16_t* arow = A  + (size_t)(m0 + r) * K;
  const bf16_t* br0  = Bt + (size_t)(n0 + r) * K;
  const bf16_t* br1  = br0 + (size_t)16 * K;
  const bf16_t* br2  = br0 + (size_t)32 * K;
  const bf16_t* br3  = br0 + (size_t)48 * K;

  v8f acc0 = {}, acc1 = {}, acc2 = {}, acc3 = {};

  for (int k0 = 0; k0 < K; k0 += 32) {
    const int ka = k0 + hv * 16;  // A: contiguous-16 layout per half-wave
    const int kb = k0 + hv * 16;  // B: contiguous-16 layout per half-wave
    v16bf af = load_afrag(arow + ka);
    v16bf bf0 = *(const v16bf*)(br0 + kb);
    v16bf bf1 = *(const v16bf*)(br1 + kb);
    v16bf bf2 = *(const v16bf*)(br2 + kb);
    v16bf bf3 = *(const v16bf*)(br3 + kb);
    acc0 = __builtin_amdgcn_wmma_f32_16x16x32_bf16(false, af, false, bf0, (short)0, acc0, false, false);
    acc1 = __builtin_amdgcn_wmma_f32_16x16x32_bf16(false, af, false, bf1, (short)0, acc1, false, false);
    acc2 = __builtin_amdgcn_wmma_f32_16x16x32_bf16(false, af, false, bf2, (short)0, acc2, false, false);
    acc3 = __builtin_amdgcn_wmma_f32_16x16x32_bf16(false, af, false, bf3, (short)0, acc3, false, false);
  }

  // C/D layout: VGPR v -> row m0 + v + hv*8 ; lane r -> column
  float* crow = C + (size_t)m0 * N + n0;
#pragma unroll
  for (int v = 0; v < 8; ++v) {
    size_t rowoff = (size_t)(v + hv * 8) * N;
    crow[rowoff +  0 + r] = acc0[v];
    crow[rowoff + 16 + r] = acc1[v];
    crow[rowoff + 32 + r] = acc2[v];
    crow[rowoff + 48 + r] = acc3[v];
  }
}

// ---------------- attention pieces ----------------

// per (node, head): dot(h[n,head,:], a_src[head,:]) and dot(..., a_dst)
__global__ void k_scores(const float* __restrict__ h, const float* __restrict__ a_src,
                         const float* __restrict__ a_dst, float* __restrict__ ssrc,
                         float* __restrict__ sdst, int Nn, int H, int D)
{
  int i = blockIdx.x * blockDim.x + threadIdx.x;
  if (i >= Nn * H) return;
  int n = i / H, hd = i - n * H;
  const float* hp = h + (size_t)n * H * D + (size_t)hd * D;
  const float* as = a_src + (size_t)hd * D;
  const float* ad = a_dst + (size_t)hd * D;
  float s0 = 0.f, s1 = 0.f;
  for (int d = 0; d < D; ++d) { float x = hp[d]; s0 += x * as[d]; s1 += x * ad[d]; }
  ssrc[i] = s0; sdst[i] = s1;
}

__device__ inline void atomicMaxF(float* addr, float val) {
  if (val >= 0.0f) atomicMax((int*)addr, __float_as_int(val));
  else             atomicMin((unsigned int*)addr, __float_as_uint(val));
}

__device__ inline void edge_endpoints(const int* __restrict__ ei, int e, int& s, int& d) {
  if (e < N_EDGES) { s = ei[e]; d = ei[N_EDGES + e]; }
  else             { s = d = e - N_EDGES; }            // self-loop
}

// e = leaky_relu(ss[src] + sd[dst]); store; segment-max over dst
__global__ void k_edge_max(const int* __restrict__ ei, const float* __restrict__ ssrc,
                           const float* __restrict__ sdst, float* __restrict__ ebuf,
                           float* __restrict__ emax, int H)
{
  int i = blockIdx.x * blockDim.x + threadIdx.x;
  if (i >= ET * H) return;
  int e = i / H, hd = i - e * H;
  int s, d; edge_endpoints(ei, e, s, d);
  float v = ssrc[s * H + hd] + sdst[d * H + hd];
  v = v > 0.f ? v : 0.2f * v;                 // leaky_relu, slope 0.2
  ebuf[i] = v;
  atomicMaxF(&emax[d * H + hd], v);
}

// ex = exp(e - emax[dst]); store in place; segment-sum into denom
__global__ void k_edge_exp(const int* __restrict__ ei, float* __restrict__ ebuf,
                           const float* __restrict__ emax, float* __restrict__ denom, int H)
{
  int i = blockIdx.x * blockDim.x + threadIdx.x;
  if (i >= ET * H) return;
  int e = i / H, hd = i - e * H;
  int s, d; edge_endpoints(ei, e, s, d);
  float ex = __expf(ebuf[i] - emax[d * H + hd]);
  ebuf[i] = ex;
  atomicAdd(&denom[d * H + hd], ex);
}

// agg[dst] += alpha * h[src] ; one block per edge, threads stride channels
__global__ void k_edge_agg(const int* __restrict__ ei, const float* __restrict__ ebuf,
                           const float* __restrict__ denom, const float* __restrict__ h,
                           float* __restrict__ agg, int H, int D)
{
  int e = blockIdx.x;
  int s, d; edge_endpoints(ei, e, s, d);
  int C = H * D;
  const float* hs = h + (size_t)s * C;
  float* ag = agg + (size_t)d * C;
  for (int c = threadIdx.x; c < C; c += blockDim.x) {
    int hd = c / D;
    float alpha = ebuf[e * H + hd] / (denom[d * H + hd] + 1e-16f);
    atomicAdd(&ag[c], alpha * hs[c]);
  }
}

__global__ void k_bias_elu_bf16(const float* __restrict__ agg, const float* __restrict__ b,
                                bf16_t* __restrict__ out, int total, int C)
{
  int i = blockIdx.x * blockDim.x + threadIdx.x;
  if (i >= total) return;
  float v = agg[i] + b[i % C];
  v = v > 0.f ? v : (__expf(v) - 1.0f);       // ELU(alpha=1)
  out[i] = (bf16_t)v;
}

__global__ void k_bias_out(const float* __restrict__ agg, const float* __restrict__ b,
                           float* __restrict__ out, int total, int C)
{
  int i = blockIdx.x * blockDim.x + threadIdx.x;
  if (i >= total) return;
  out[i] = agg[i] + b[i % C];
}

// ---------------- host-side orchestration ----------------

extern "C" void kernel_launch(void* const* d_in, const int* in_sizes, int n_in,
                              void* d_out, int out_size, void* d_ws, size_t ws_size,
                              hipStream_t stream)
{
  const float* x      = (const float*)d_in[0];
  const int*   ei     = (const int*)  d_in[1];   // [2, E] flat
  const float* W1     = (const float*)d_in[2];   // [768, 2048]
  const float* a1_src = (const float*)d_in[3];   // [4, 512]
  const float* a1_dst = (const float*)d_in[4];
  const float* b1     = (const float*)d_in[5];   // [2048]
  const float* W2     = (const float*)d_in[6];   // [2048, 768]
  const float* a2_src = (const float*)d_in[7];   // [1, 768]
  const float* a2_dst = (const float*)d_in[8];
  const float* b2     = (const float*)d_in[9];   // [768]
  float* out = (float*)d_out;                    // [10000, 768]

  char* ws = (char*)d_ws;
  size_t off = 0;
  auto alloc = [&](size_t bytes) -> char* {
    char* p = ws + off;
    off += bytes;
    off = (off + 255) & ~(size_t)255;
    return p;
  };

  bf16_t* xb    = (bf16_t*)alloc((size_t)N_NODES * IN_DIM  * 2);
  bf16_t* w1t   = (bf16_t*)alloc((size_t)HDC     * IN_DIM  * 2);  // [2048][768]
  bf16_t* w2t   = (bf16_t*)alloc((size_t)OUT_DIM * HDC     * 2);  // [768][2048]
  float*  h1    = (float*) alloc((size_t)N_NODES * HDC     * 4);  // reused as bf16 h1b later
  float*  aggA  = (float*) alloc((size_t)N_NODES * HDC     * 4);  // agg1; reused for h2+agg2
  float*  ss1   = (float*) alloc((size_t)N_NODES * HEADS * 4);
  float*  sd1   = (float*) alloc((size_t)N_NODES * HEADS * 4);
  float*  emax1 = (float*) alloc((size_t)N_NODES * HEADS * 4);
  float*  den1  = (float*) alloc((size_t)N_NODES * HEADS * 4);
  float*  e1    = (float*) alloc((size_t)ET * HEADS * 4);
  float*  ss2   = (float*) alloc((size_t)N_NODES * 4);
  float*  sd2   = (float*) alloc((size_t)N_NODES * 4);
  float*  emax2 = (float*) alloc((size_t)N_NODES * 4);
  float*  den2  = (float*) alloc((size_t)N_NODES * 4);
  float*  e2    = (float*) alloc((size_t)ET * 4);

  bf16_t* h1b  = (bf16_t*)h1;                          // overlay: h1 dead after agg1
  float*  h2   = aggA;                                 // overlay: agg1 dead after ELU
  float*  agg2 = aggA + (size_t)N_NODES * HDC / 2;     // +40.96MB, past h2's 30.72MB

  const int T = 256;
  // precision conversion + weight transposes
  k_f32_to_bf16<<<(N_NODES * IN_DIM + T - 1) / T, T, 0, stream>>>(x, xb, N_NODES * IN_DIM);
  k_transpose_bf16<<<(IN_DIM * HDC + T - 1) / T, T, 0, stream>>>(W1, w1t, IN_DIM, HDC);
  k_transpose_bf16<<<(HDC * OUT_DIM + T - 1) / T, T, 0, stream>>>(W2, w2t, HDC, OUT_DIM);

  // -------- layer 1 --------
  k_gemm_bf16_wmma<<<dim3(N_NODES / 16, HDC / 64), 32, 0, stream>>>(xb, w1t, h1,
                                                                    N_NODES, HDC, IN_DIM);
  k_scores<<<(N_NODES * HEADS + 127) / 128, 128, 0, stream>>>(h1, a1_src, a1_dst,
                                                              ss1, sd1, N_NODES, HEADS, HID);
  k_fill<<<(N_NODES * HEADS + T - 1) / T, T, 0, stream>>>(emax1, -3.0e38f, N_NODES * HEADS);
  k_fill<<<(N_NODES * HEADS + T - 1) / T, T, 0, stream>>>(den1, 0.0f, N_NODES * HEADS);
  k_fill<<<(N_NODES * HDC + T - 1) / T, T, 0, stream>>>(aggA, 0.0f, N_NODES * HDC);
  k_edge_max<<<(ET * HEADS + T - 1) / T, T, 0, stream>>>(ei, ss1, sd1, e1, emax1, HEADS);
  k_edge_exp<<<(ET * HEADS + T - 1) / T, T, 0, stream>>>(ei, e1, emax1, den1, HEADS);
  k_edge_agg<<<ET, 256, 0, stream>>>(ei, e1, den1, h1, aggA, HEADS, HID);
  k_bias_elu_bf16<<<(N_NODES * HDC + T - 1) / T, T, 0, stream>>>(aggA, b1, h1b,
                                                                 N_NODES * HDC, HDC);

  // -------- layer 2 --------
  k_gemm_bf16_wmma<<<dim3(N_NODES / 16, OUT_DIM / 64), 32, 0, stream>>>(h1b, w2t, h2,
                                                                        N_NODES, OUT_DIM, HDC);
  k_scores<<<(N_NODES + 127) / 128, 128, 0, stream>>>(h2, a2_src, a2_dst,
                                                      ss2, sd2, N_NODES, 1, OUT_DIM);
  k_fill<<<(N_NODES + T - 1) / T, T, 0, stream>>>(emax2, -3.0e38f, N_NODES);
  k_fill<<<(N_NODES + T - 1) / T, T, 0, stream>>>(den2, 0.0f, N_NODES);
  k_fill<<<(N_NODES * OUT_DIM + T - 1) / T, T, 0, stream>>>(agg2, 0.0f, N_NODES * OUT_DIM);
  k_edge_max<<<(ET + T - 1) / T, T, 0, stream>>>(ei, ss2, sd2, e2, emax2, 1);
  k_edge_exp<<<(ET + T - 1) / T, T, 0, stream>>>(ei, e2, emax2, den2, 1);
  k_edge_agg<<<ET, 256, 0, stream>>>(ei, e2, den2, h2, agg2, 1, OUT_DIM);
  k_bias_out<<<(N_NODES * OUT_DIM + T - 1) / T, T, 0, stream>>>(agg2, b2, out,
                                                                N_NODES * OUT_DIM, OUT_DIM);
}
